// Swin_b_Deep_4320737099926
// MI455X (gfx1250) — compile-verified, matrix operand order checked by reading, not run on hardware
//
#include <hip/hip_runtime.h>
#include <hip/hip_bf16.h>
#include <math.h>

typedef __bf16 bf16_t;
typedef __attribute__((ext_vector_type(16))) __bf16 v16bf;
typedef __attribute__((ext_vector_type(8)))  float  v8f;

union FragU { v16bf v; uint4 u[2]; };

__device__ inline v8f wmma_bf16(v16bf a, v16bf b, v8f c) {
  return __builtin_amdgcn_wmma_f32_16x16x32_bf16(false, a, false, b, (short)0, c, false, false);
}

// Load one 16x(K=32) bf16 fragment. Works for A-rows or B-columns (W^T rows).
// CDNA5 16-bit A layout: lanes 0-15 -> rows, halves 0-7 = K k0..k0+7, halves 8-15 = K k0+16..k0+23
// with k0 = (lane>=16 ? 8 : 0).
__device__ inline v16bf load_frag(const bf16_t* __restrict__ base, long row0, int ld,
                                  int kBase, long rowMax) {
  int lane = threadIdx.x & 31;
  int sub  = lane >> 4;
  int r    = lane & 15;
  long row = row0 + r;
  int  kA  = kBase + sub * 8;
  FragU f;
  if (row < rowMax) {
    const uint4* p = reinterpret_cast<const uint4*>(base + row * (long)ld + kA);
    f.u[0] = p[0];   // halves kA .. kA+7
    f.u[1] = p[2];   // halves kA+16 .. kA+23
  } else {
    f.u[0] = make_uint4(0u,0u,0u,0u);
    f.u[1] = make_uint4(0u,0u,0u,0u);
  }
  return f.v;
}

// ---------------- GEMM with fused epilogues ----------------
enum { EPI_F32 = 0, EPI_GELU = 1, EPI_RESADD = 2, EPI_QKV = 3, EPI_UNWIN = 4 };

struct EpiArgs {
  float*  outF;
  bf16_t* outB;
  const float* bias;
  bf16_t* q; bf16_t* k; bf16_t* vt;
  int heads; int C; int shift; int H; int nh;
};

template <int MODE>
__device__ inline void epi_store(const v8f& c, long tm, long tn, int M, int N, const EpiArgs& e) {
  int lane = threadIdx.x & 31;
  int sub  = lane >> 4;
  int r    = lane & 15;
  long col = tn + r;
  if (col >= N) return;
  float bv = e.bias ? e.bias[col] : 0.0f;
#pragma unroll
  for (int j = 0; j < 8; ++j) {
    long row = tm + sub * 8 + j;
    if (row >= M) continue;
    float v = c[j] + bv;
    if constexpr (MODE == EPI_F32) {
      e.outF[row * (long)N + col] = v;
    } else if constexpr (MODE == EPI_GELU) {
      float g = 0.5f * v * (1.0f + erff(v * 0.70710678118654752f));
      e.outB[row * (long)N + col] = (bf16_t)g;
    } else if constexpr (MODE == EPI_RESADD) {
      e.outF[row * (long)N + col] += v;
    } else if constexpr (MODE == EPI_QKV) {
      int C = e.C;
      int which = (int)(col / C);
      int cc = (int)(col - (long)which * C);
      int hh = cc >> 5, f = cc & 31;
      long win = row >> 6;
      int  rr  = (int)(row & 63);
      long base = win * (long)e.heads + hh;
      if (which == 0)      e.q [(base * 64 + rr) * 32 + f] = (bf16_t)(v * 0.17677669529663687f);
      else if (which == 1) e.k [(base * 64 + rr) * 32 + f] = (bf16_t)v;
      else                 e.vt[(base * 32 + f) * 64 + rr] = (bf16_t)v;
    } else if constexpr (MODE == EPI_UNWIN) {
      long win = row >> 6;
      int  rr  = (int)(row & 63);
      if (rr < 49) {
        int wpi = e.nh * e.nh;
        int b  = (int)(win / wpi);
        int wl = (int)(win % wpi);
        int wh = wl / e.nh, ww = wl % e.nh;
        int y = wh * 7 + rr / 7, x = ww * 7 + rr % 7;
        if (e.shift) { y = (y + e.shift) % e.H; x = (x + e.shift) % e.H; }
        long t = ((long)b * e.H + y) * e.H + x;
        e.outF[t * (long)N + col] += v;   // residual add, unique (row,col)->element
      }
    }
  }
}

// C = A(MxK) * W^T (W is NxK, row-major) ; bf16 inputs, f32 accumulate.
// Block: 256 thr = 8 waves, 2x4 wave grid, each wave a 32x32 tile (2x2 WMMA tiles).
template <int MODE>
__global__ __launch_bounds__(256) void gemm_bf16(const bf16_t* __restrict__ A,
                                                 const bf16_t* __restrict__ W,
                                                 int M, int N, int K, EpiArgs e) {
  int wid = threadIdx.x >> 5;
  int wm = wid >> 2;      // 0..1
  int wn = wid & 3;       // 0..3
  long m0 = (long)blockIdx.y * 64 + wm * 32;
  long n0 = (long)blockIdx.x * 128 + wn * 32;
  v8f c00 = {0,0,0,0,0,0,0,0}, c01 = {0,0,0,0,0,0,0,0};
  v8f c10 = {0,0,0,0,0,0,0,0}, c11 = {0,0,0,0,0,0,0,0};
  for (int kk = 0; kk < K; kk += 32) {
    v16bf a0 = load_frag(A, m0,      K, kk, M);
    v16bf a1 = load_frag(A, m0 + 16, K, kk, M);
    v16bf b0 = load_frag(W, n0,      K, kk, N);
    v16bf b1 = load_frag(W, n0 + 16, K, kk, N);
    c00 = wmma_bf16(a0, b0, c00);
    c01 = wmma_bf16(a0, b1, c01);
    c10 = wmma_bf16(a1, b0, c10);
    c11 = wmma_bf16(a1, b1, c11);
  }
  epi_store<MODE>(c00, m0,      n0,      M, N, e);
  epi_store<MODE>(c01, m0,      n0 + 16, M, N, e);
  epi_store<MODE>(c10, m0 + 16, n0,      M, N, e);
  epi_store<MODE>(c11, m0 + 16, n0 + 16, M, N, e);
}

// ---------------- fused attention: S = QK^T, +bias+mask, softmax, O = P V ----------------
__device__ inline int reg3(int y, int H) { return (y < H - 7) ? 0 : ((y < H - 3) ? 1 : 2); }

__global__ __launch_bounds__(128) void attn_kernel(const bf16_t* __restrict__ q,
                                                   const bf16_t* __restrict__ k,
                                                   const bf16_t* __restrict__ vt,
                                                   const float* __restrict__ rel_bias,
                                                   bf16_t* __restrict__ attnout,
                                                   int heads, int C, int shift, int nh, int H) {
  __shared__ float Ssm[64 * 64];
  __shared__ __align__(16) bf16_t Psm[64 * 64];
  int id  = blockIdx.x;
  int win = id / heads;
  int h   = id - win * heads;
  const bf16_t* qb = q  + (long)(win * heads + h) * 2048;
  const bf16_t* kb = k  + (long)(win * heads + h) * 2048;
  const bf16_t* vb = vt + (long)(win * heads + h) * 2048;
  int wid  = threadIdx.x >> 5;
  int lane = threadIdx.x & 31;
  int sub  = lane >> 4;
  int r    = lane & 15;

  // scores: wave `wid` computes rows [16*wid, 16*wid+16), all 64 cols; K=32 -> one WMMA/tile
  {
    v16bf a = load_frag(qb, 16 * wid, 32, 0, 64);
#pragma unroll
    for (int jt = 0; jt < 4; ++jt) {
      v16bf b = load_frag(kb, 16 * jt, 32, 0, 64);
      v8f c = {0,0,0,0,0,0,0,0};
      c = wmma_bf16(a, b, c);
#pragma unroll
      for (int jj = 0; jj < 8; ++jj)
        Ssm[(16 * wid + sub * 8 + jj) * 64 + 16 * jt + r] = c[jj];
    }
  }
  __syncthreads();

  int t = threadIdx.x;
  if (t < 64) {
    if (t < 49) {
      int yi = t / 7, xi = t % 7;
      int wpi = nh * nh;
      int wl = win % wpi;
      int wh = wl / nh, ww = wl % nh;
      int idI = 0;
      if (shift) idI = reg3(wh * 7 + yi, H) * 3 + reg3(ww * 7 + xi, H);
      float mx = -1e30f;
      for (int c = 0; c < 49; ++c) {
        int yj = c / 7, xj = c % 7;
        int ridx = (yi - yj + 6) * 13 + (xi - xj + 6);
        float val = Ssm[t * 64 + c] + rel_bias[ridx * heads + h];
        if (shift) {
          int idJ = reg3(wh * 7 + yj, H) * 3 + reg3(ww * 7 + xj, H);
          if (idJ != idI) val -= 100.0f;
        }
        Ssm[t * 64 + c] = val;
        mx = fmaxf(mx, val);
      }
      float sum = 0.0f;
      for (int c = 0; c < 49; ++c) {
        float ev = __expf(Ssm[t * 64 + c] - mx);
        Ssm[t * 64 + c] = ev;
        sum += ev;
      }
      float inv = 1.0f / sum;
      for (int c = 0; c < 49; ++c) Psm[t * 64 + c] = (bf16_t)(Ssm[t * 64 + c] * inv);
      for (int c = 49; c < 64; ++c) Psm[t * 64 + c] = (bf16_t)0.0f;
    } else {
      for (int c = 0; c < 64; ++c) Psm[t * 64 + c] = (bf16_t)0.0f;
    }
  }
  __syncthreads();

  // O = P(64x64) @ V(64x32) via Vt rows; wave `wid` -> rows [16*wid,16*wid+16), 2 col tiles, K=64
  v8f o0 = {0,0,0,0,0,0,0,0}, o1 = {0,0,0,0,0,0,0,0};
#pragma unroll
  for (int ks = 0; ks < 2; ++ks) {
    v16bf a  = load_frag((const bf16_t*)Psm, 16 * wid, 64, ks * 32, 64);
    v16bf b0 = load_frag(vb, 0,  64, ks * 32, 32);
    v16bf b1 = load_frag(vb, 16, 64, ks * 32, 32);
    o0 = wmma_bf16(a, b0, o0);
    o1 = wmma_bf16(a, b1, o1);
  }
#pragma unroll
  for (int jj = 0; jj < 8; ++jj) {
    long gr = (long)win * 64 + 16 * wid + sub * 8 + jj;
    attnout[gr * C + h * 32 + 0  + r] = (bf16_t)o0[jj];
    attnout[gr * C + h * 32 + 16 + r] = (bf16_t)o1[jj];
  }
}

// ---------------- LN / window / merge / misc ----------------
__global__ __launch_bounds__(128) void ln_window(const float* __restrict__ x,
                                                 const float* __restrict__ g,
                                                 const float* __restrict__ b,
                                                 bf16_t* __restrict__ winbuf,
                                                 int H, int C, int shift, int nh) {
  long tok = blockIdx.x;
  long win = tok >> 6;
  int  r   = (int)(tok & 63);
  bf16_t* out = winbuf + tok * C;
  int tid = threadIdx.x;
  if (r >= 49) {
    for (int c = tid; c < C; c += 128) out[c] = (bf16_t)0.0f;
    return;
  }
  int wpi = nh * nh;
  int bimg = (int)(win / wpi);
  int wl = (int)(win % wpi);
  int wh = wl / nh, ww = wl % nh;
  int y = wh * 7 + r / 7, xx = ww * 7 + r % 7;
  if (shift) { y = (y + shift) % H; xx = (xx + shift) % H; }
  const float* row = x + ((long)(bimg * H + y) * H + xx) * C;
  __shared__ float s1[128], s2[128];
  float a = 0.0f, a2 = 0.0f;
  for (int c = tid; c < C; c += 128) { float v = row[c]; a += v; a2 += v * v; }
  s1[tid] = a; s2[tid] = a2; __syncthreads();
  for (int off = 64; off > 0; off >>= 1) {
    if (tid < off) { s1[tid] += s1[tid + off]; s2[tid] += s2[tid + off]; }
    __syncthreads();
  }
  float mean = s1[0] / C;
  float var  = s2[0] / C - mean * mean;
  float rs = rsqrtf(var + 1e-5f);
  for (int c = tid; c < C; c += 128)
    out[c] = (bf16_t)((row[c] - mean) * rs * g[c] + b[c]);
}

__global__ __launch_bounds__(256) void ln_kernel(const float* __restrict__ x,
                                                 const float* __restrict__ g,
                                                 const float* __restrict__ b,
                                                 float* outF, bf16_t* outB, int C) {
  long t = blockIdx.x;
  const float* row = x + t * (long)C;
  int tid = threadIdx.x;
  __shared__ float s1[256], s2[256];
  float a = 0.0f, a2 = 0.0f;
  for (int c = tid; c < C; c += 256) { float v = row[c]; a += v; a2 += v * v; }
  s1[tid] = a; s2[tid] = a2; __syncthreads();
  for (int off = 128; off > 0; off >>= 1) {
    if (tid < off) { s1[tid] += s1[tid + off]; s2[tid] += s2[tid + off]; }
    __syncthreads();
  }
  float mean = s1[0] / C;
  float var  = s2[0] / C - mean * mean;
  float rs = rsqrtf(var + 1e-5f);
  for (int c = tid; c < C; c += 256) {
    float v = (row[c] - mean) * rs * g[c] + b[c];
    if (outF) outF[t * (long)C + c] = v;
    if (outB) outB[t * (long)C + c] = (bf16_t)v;
  }
}

__global__ __launch_bounds__(256) void merge_ln(const float* __restrict__ x,
                                                const float* __restrict__ g,
                                                const float* __restrict__ b,
                                                bf16_t* __restrict__ out,
                                                int H, int C) {
  int H2 = H / 2;
  long t = blockIdx.x;
  int bimg = (int)(t / (H2 * H2));
  int rem  = (int)(t % (H2 * H2));
  int i = rem / H2, j = rem % H2;
  int C4 = 4 * C;
  int tid = threadIdx.x;
  __shared__ float buf[2048];
  __shared__ float s1[256], s2[256];
  float a = 0.0f, a2 = 0.0f;
  for (int idx = tid; idx < C4; idx += 256) {
    int qd = idx / C, c = idx % C;
    int di = qd & 1, dj = qd >> 1;
    float v = x[(((long)bimg * H + 2 * i + di) * H + 2 * j + dj) * C + c];
    buf[idx] = v; a += v; a2 += v * v;
  }
  s1[tid] = a; s2[tid] = a2; __syncthreads();
  for (int off = 128; off > 0; off >>= 1) {
    if (tid < off) { s1[tid] += s1[tid + off]; s2[tid] += s2[tid + off]; }
    __syncthreads();
  }
  float mean = s1[0] / C4;
  float var  = s2[0] / C4 - mean * mean;
  float rs = rsqrtf(var + 1e-5f);
  for (int idx = tid; idx < C4; idx += 256)
    out[t * (long)C4 + idx] = (bf16_t)((buf[idx] - mean) * rs * g[idx] + b[idx]);
}

__global__ void cvt_f32_bf16(const float* __restrict__ s, bf16_t* __restrict__ d, long n) {
  long i = (long)blockIdx.x * blockDim.x + threadIdx.x;
  if (i < n) d[i] = (bf16_t)s[i];
}

__global__ void patch_gather(const float* __restrict__ x, bf16_t* __restrict__ out) {
  long i = (long)blockIdx.x * blockDim.x + threadIdx.x;
  if (i >= 25088L * 64) return;
  long t = i >> 6;
  int kk = (int)(i & 63);
  if (kk >= 48) { out[i] = (bf16_t)0.0f; return; }
  int b = (int)(t / 3136), rem = (int)(t % 3136);
  int ph = rem / 56, pw = rem % 56;
  int c = kk >> 4, r2 = kk & 15, ii = r2 >> 2, jj = r2 & 3;
  out[i] = (bf16_t)x[(((long)b * 3 + c) * 224 + 4 * ph + ii) * 224 + 4 * pw + jj];
}

__global__ void patchw_cvt(const float* __restrict__ w, bf16_t* __restrict__ d) {
  int i = blockIdx.x * blockDim.x + threadIdx.x;
  if (i >= 128 * 64) return;
  int o = i >> 6, kk = i & 63;
  d[i] = (kk < 48) ? (bf16_t)w[o * 48 + kk] : (bf16_t)0.0f;
}

__global__ void pool_kernel(const float* __restrict__ x, float* __restrict__ out, int C) {
  int i = blockIdx.x * blockDim.x + threadIdx.x;
  if (i >= 8 * C) return;
  int b = i / C, c = i % C;
  float s = 0.0f;
  for (int p = 0; p < 49; ++p) s += x[((long)b * 49 + p) * C + c];
  out[i] = s / 49.0f;
}

template <int MODE>  // 0: leaky_relu + BN eval ; 1: plain
__global__ void head_fc(const float* __restrict__ in, const float* __restrict__ w,
                        const float* __restrict__ bias,
                        const float* bn_g, const float* bn_b, const float* bn_m, const float* bn_v,
                        float* __restrict__ out, int B, int N, int K) {
  int i = blockIdx.x * blockDim.x + threadIdx.x;
  if (i >= B * N) return;
  int b = i / N, n = i % N;
  const float* wr = w + (long)n * K;
  const float* xr = in + (long)b * K;
  float s = bias[n];
  for (int kk = 0; kk < K; ++kk) s += xr[kk] * wr[kk];
  if constexpr (MODE == 0) {
    s = (s > 0.0f) ? s : 0.01f * s;
    s = (s - bn_m[n]) * rsqrtf(bn_v[n] + 1e-5f) * bn_g[n] + bn_b[n];
  }
  out[i] = s;
}

// ---------------- host ----------------
static void gemm_dispatch(int mode, hipStream_t s, const bf16_t* A, const bf16_t* W,
                          long M, int N, int K, const EpiArgs& e) {
  dim3 g((N + 127) / 128, (unsigned)((M + 63) / 64));
  switch (mode) {
    case EPI_F32:    gemm_bf16<EPI_F32>   <<<g, 256, 0, s>>>(A, W, (int)M, N, K, e); break;
    case EPI_GELU:   gemm_bf16<EPI_GELU>  <<<g, 256, 0, s>>>(A, W, (int)M, N, K, e); break;
    case EPI_RESADD: gemm_bf16<EPI_RESADD><<<g, 256, 0, s>>>(A, W, (int)M, N, K, e); break;
    case EPI_QKV:    gemm_bf16<EPI_QKV>   <<<g, 256, 0, s>>>(A, W, (int)M, N, K, e); break;
    case EPI_UNWIN:  gemm_bf16<EPI_UNWIN> <<<g, 256, 0, s>>>(A, W, (int)M, N, K, e); break;
  }
}

extern "C" void kernel_launch(void* const* d_in, const int* in_sizes, int n_in,
                              void* d_out, int out_size, void* d_ws, size_t ws_size,
                              hipStream_t stream) {
  (void)in_sizes; (void)n_in; (void)out_size; (void)ws_size;
  int ip = 0;
  auto nx = [&]() { return (const float*)d_in[ip++]; };

  const float* x_in = nx();
  const float* pw = nx(); const float* pb = nx();
  const float* plg = nx(); const float* plb = nx();

  char* wsp = (char*)d_ws;
  size_t off = 0;
  auto carve = [&](size_t bytes) -> void* {
    void* p = wsp + off;
    off = (off + bytes + 255) & ~(size_t)255;
    return p;
  };
  float*  xbuf    = (float*)carve(3211264ULL * 4);   // f32 feature map (max 25088x128)
  bf16_t* lnbuf   = (bf16_t*)carve(3211264ULL * 2);  // bf16 LN / patch buffer
  bf16_t* winbuf  = (bf16_t*)carve(4194304ULL * 2);  // windowed LN tokens
  bf16_t* qbuf    = (bf16_t*)carve(4194304ULL * 2);
  bf16_t* kbuf    = (bf16_t*)carve(4194304ULL * 2);
  bf16_t* vtbuf   = (bf16_t*)carve(4194304ULL * 2);
  bf16_t* attnout = (bf16_t*)carve(4194304ULL * 2);
  bf16_t* hbuf    = (bf16_t*)carve(12845056ULL * 2); // MLP hidden / merge buffer
  bf16_t* wbuf    = (bf16_t*)carve(4194304ULL * 2);  // bf16 weight staging
  float*  pooled  = (float*)carve(8192 * 4);
  float*  h1      = (float*)carve(4096 * 4);
  float*  h2      = (float*)carve(1024 * 4);

  // --- patch embed: conv4x4 s4 as GEMM (K=48 padded to 64) + bias, then LN ---
  patch_gather<<<dim3((unsigned)((25088L * 64 + 255) / 256)), 256, 0, stream>>>(x_in, lnbuf);
  patchw_cvt<<<dim3((128 * 64 + 255) / 256), 256, 0, stream>>>(pw, wbuf);
  {
    EpiArgs e{}; e.outF = xbuf; e.bias = pb;
    gemm_dispatch(EPI_F32, stream, lnbuf, wbuf, 25088, 128, 64, e);
  }
  ln_kernel<<<dim3(25088), 256, 0, stream>>>(xbuf, plg, plb, xbuf, nullptr, 128);

  const int DEPTH[4] = {2, 2, 18, 2};
  const int HEADSA[4] = {4, 8, 16, 32};
  int H = 56, C = 128;
  for (int si = 0; si < 4; ++si) {
    int heads = HEADSA[si];
    int nh = H / 7;
    int nwin = 8 * nh * nh;
    long Mw = (long)nwin * 64;
    long Mt = 8L * H * H;
    for (int bi = 0; bi < DEPTH[si]; ++bi) {
      const float* ln1g = nx(); const float* ln1b = nx();
      const float* qkvw = nx(); const float* qkvb = nx();
      const float* relb = nx();
      const float* projw = nx(); const float* projb = nx();
      const float* ln2g = nx(); const float* ln2b = nx();
      const float* fc1w = nx(); const float* fc1b = nx();
      const float* fc2w = nx(); const float* fc2b = nx();
      int shift = ((bi & 1) && H > 7) ? 3 : 0;

      // attention
      ln_window<<<dim3((unsigned)Mw), 128, 0, stream>>>(xbuf, ln1g, ln1b, winbuf, H, C, shift, nh);
      cvt_f32_bf16<<<dim3((unsigned)((3L * C * C + 255) / 256)), 256, 0, stream>>>(qkvw, wbuf, 3L * C * C);
      {
        EpiArgs e{}; e.bias = qkvb; e.q = qbuf; e.k = kbuf; e.vt = vtbuf; e.heads = heads; e.C = C;
        gemm_dispatch(EPI_QKV, stream, winbuf, wbuf, Mw, 3 * C, C, e);
      }
      attn_kernel<<<dim3((unsigned)(nwin * heads)), 128, 0, stream>>>(
          qbuf, kbuf, vtbuf, relb, attnout, heads, C, shift, nh, H);
      cvt_f32_bf16<<<dim3((unsigned)(((long)C * C + 255) / 256)), 256, 0, stream>>>(projw, wbuf, (long)C * C);
      {
        EpiArgs e{}; e.outF = xbuf; e.bias = projb; e.shift = shift; e.H = H; e.nh = nh;
        gemm_dispatch(EPI_UNWIN, stream, attnout, wbuf, Mw, C, C, e);
      }

      // MLP
      ln_kernel<<<dim3((unsigned)Mt), 256, 0, stream>>>(xbuf, ln2g, ln2b, nullptr, lnbuf, C);
      cvt_f32_bf16<<<dim3((unsigned)((4L * C * C + 255) / 256)), 256, 0, stream>>>(fc1w, wbuf, 4L * C * C);
      {
        EpiArgs e{}; e.outB = hbuf; e.bias = fc1b;
        gemm_dispatch(EPI_GELU, stream, lnbuf, wbuf, Mt, 4 * C, C, e);
      }
      cvt_f32_bf16<<<dim3((unsigned)((4L * C * C + 255) / 256)), 256, 0, stream>>>(fc2w, wbuf, 4L * C * C);
      {
        EpiArgs e{}; e.outF = xbuf; e.bias = fc2b;
        gemm_dispatch(EPI_RESADD, stream, hbuf, wbuf, Mt, C, 4 * C, e);
      }
    }
    if (si < 3) {
      const float* mlg = nx(); const float* mlb = nx(); const float* mw = nx();
      long M2 = 8L * (H / 2) * (H / 2);
      merge_ln<<<dim3((unsigned)M2), 256, 0, stream>>>(xbuf, mlg, mlb, hbuf, H, C);
      cvt_f32_bf16<<<dim3((unsigned)((8L * C * C + 255) / 256)), 256, 0, stream>>>(mw, wbuf, 8L * C * C);
      {
        EpiArgs e{}; e.outF = xbuf;
        gemm_dispatch(EPI_F32, stream, hbuf, wbuf, M2, 2 * C, 4 * C, e);
      }
      H /= 2; C *= 2;
    }
  }

  // final LN, pool, head MLP
  const float* flg = nx(); const float* flb = nx();
  ln_kernel<<<dim3(392), 256, 0, stream>>>(xbuf, flg, flb, xbuf, nullptr, 1024);
  pool_kernel<<<dim3(32), 256, 0, stream>>>(xbuf, pooled, 1024);

  const float* w1 = nx(); const float* b1 = nx();
  const float* g1 = nx(); const float* bb1 = nx(); const float* m1 = nx(); const float* v1 = nx();
  const float* w2 = nx(); const float* b2 = nx();
  const float* g2 = nx(); const float* bb2 = nx(); const float* m2 = nx(); const float* v2 = nx();
  const float* w3 = nx(); const float* b3 = nx();
  head_fc<0><<<dim3((8 * 512 + 127) / 128), 128, 0, stream>>>(pooled, w1, b1, g1, bb1, m1, v1, h1, 8, 512, 1024);
  head_fc<0><<<dim3((8 * 128 + 127) / 128), 128, 0, stream>>>(h1, w2, b2, g2, bb2, m2, v2, h2, 8, 128, 512);
  head_fc<1><<<dim3(2), 128, 0, stream>>>(h2, w3, b3, nullptr, nullptr, nullptr, nullptr,
                                          (float*)d_out, 8, 18, 128);
}
